// AtomicLinear_32555852104251
// MI455X (gfx1250) — matrix operand order repaired and checked by reading, not compiled, not run on hardware
//
#include <hip/hip_runtime.h>
#include <hip/hip_bf16.h>

// Problem sizes (from reference)
#define IN_F   2048
#define OUT_F  2048
#define BATCHN 4096

#define TILE_K 32
#define LSTR   40            // 32 K-elems + 8 pad (bf16) -> 80B row stride, conflict-free
#define NKT    (IN_F / TILE_K)

typedef __attribute__((ext_vector_type(4)))  unsigned short v4us;
typedef __attribute__((ext_vector_type(8)))  unsigned short v8us;
typedef __attribute__((ext_vector_type(8)))  __bf16 v8bf;
typedef __attribute__((ext_vector_type(16))) __bf16 v16bf;
typedef __attribute__((ext_vector_type(8)))  float  v8f;

// float -> bf16 (round to nearest even), raw bits
__device__ __forceinline__ unsigned short f2bf(float f) {
  unsigned u = __float_as_uint(f);
  u += 0x7FFFu + ((u >> 16) & 1u);
  return (unsigned short)(u >> 16);
}
__device__ __forceinline__ float bf2f(unsigned short h) {
  return __uint_as_float(((unsigned)h) << 16);
}

// Build a 16-element bf16 fragment from two contiguous 16-byte LDS chunks.
__device__ __forceinline__ v16bf ld_frag2(const unsigned short* p0,
                                          const unsigned short* p1) {
  v8bf a = *(const v8bf*)p0;
  v8bf b = *(const v8bf*)p1;
  v16bf r;
#pragma unroll
  for (int i = 0; i < 8; ++i) { r[i] = a[i]; r[i + 8] = b[i]; }
  return r;
}

// ---------------- Pass 1: split fp32 -> bf16 hi + bf16 lo (one-shot) -------
__global__ __launch_bounds__(256)
void split_pass(const float* __restrict__ src,
                unsigned short* __restrict__ hi,
                unsigned short* __restrict__ lo, int n4) {
  const int i = blockIdx.x * 256 + threadIdx.x;
  if (i >= n4) return;
  float4 v = ((const float4*)src)[i];
  float f[4] = {v.x, v.y, v.z, v.w};
  v4us h, l;
#pragma unroll
  for (int j = 0; j < 4; ++j) {
    unsigned short hh = f2bf(f[j]);
    h[j] = hh;
    l[j] = f2bf(f[j] - bf2f(hh));
  }
  ((v4us*)hi)[i] = h;
  ((v4us*)lo)[i] = l;
}

// ---------------- Pass 2: split-bf16 WMMA GEMM, double-buffered LDS --------
__global__ __launch_bounds__(256)
void gemm_split(const unsigned short* __restrict__ Xhi,
                const unsigned short* __restrict__ Xlo,
                const unsigned short* __restrict__ Whi,
                const unsigned short* __restrict__ Wlo,
                const float* __restrict__ Bv,
                float* __restrict__ O) {
  // [buffer][array: Xhi,Xlo,Whi,Wlo][128 rows x LSTR]  = 80 KB
  __shared__ unsigned short sb[2][4][128 * LSTR];

  const int bn = blockIdx.x;     // output-feature tile (128), 16 tiles
  const int bm = blockIdx.y;     // batch tile (128), 32 tiles
  const int t    = threadIdx.x;  // 8 wave32
  const int lane = t & 31;
  const int wave = t >> 5;
  const int wr = wave & 1;       // 2 row groups of 64
  const int wc = wave >> 1;      // 4 col groups of 32
  const int lm = lane & 15;
  const int lh = lane >> 4;

  // staging: thread -> (row, 16-elem K chunk)
  const int srow = t >> 1;
  const int scol = (t & 1) * 16;
  const int so   = srow * LSTR + scol;
  const size_t xoff = (size_t)(bm * 128 + srow) * IN_F + scol;
  const size_t woff = (size_t)(bn * 128 + srow) * IN_F + scol;
  const unsigned short* gXh = Xhi + xoff;
  const unsigned short* gXl = Xlo + xoff;
  const unsigned short* gWh = Whi + woff;
  const unsigned short* gWl = Wlo + woff;

  v8f acc[4][2];
#pragma unroll
  for (int mi = 0; mi < 4; ++mi)
#pragma unroll
    for (int ni = 0; ni < 2; ++ni)
#pragma unroll
      for (int r = 0; r < 8; ++r) acc[mi][ni][r] = 0.0f;

  // prologue: stage tile 0 into buffer 0
  *(v8us*)&sb[0][0][so]     = *(const v8us*)(gXh);
  *(v8us*)&sb[0][0][so + 8] = *(const v8us*)(gXh + 8);
  *(v8us*)&sb[0][1][so]     = *(const v8us*)(gXl);
  *(v8us*)&sb[0][1][so + 8] = *(const v8us*)(gXl + 8);
  *(v8us*)&sb[0][2][so]     = *(const v8us*)(gWh);
  *(v8us*)&sb[0][2][so + 8] = *(const v8us*)(gWh + 8);
  *(v8us*)&sb[0][3][so]     = *(const v8us*)(gWl);
  *(v8us*)&sb[0][3][so + 8] = *(const v8us*)(gWl + 8);
  __syncthreads();

  int buf = 0;
  for (int kt = 0; kt < NKT; ++kt) {
    // ---- issue next tile's global loads before compute (latency overlap) ----
    v8us st[8];
    const bool more = (kt + 1) < NKT;
    if (more) {
      const int k0n = (kt + 1) * TILE_K;
      st[0] = *(const v8us*)(gXh + k0n);
      st[1] = *(const v8us*)(gXh + k0n + 8);
      st[2] = *(const v8us*)(gXl + k0n);
      st[3] = *(const v8us*)(gXl + k0n + 8);
      st[4] = *(const v8us*)(gWh + k0n);
      st[5] = *(const v8us*)(gWh + k0n + 8);
      st[6] = *(const v8us*)(gWl + k0n);
      st[7] = *(const v8us*)(gWl + k0n + 8);
      if (kt + 2 < NKT) {  // gfx1250 global_prefetch_b8
        __builtin_prefetch(gXh + (kt + 2) * TILE_K, 0, 0);
        __builtin_prefetch(gWh + (kt + 2) * TILE_K, 0, 0);
      }
    }

    const unsigned short* bXh = sb[buf][0];
    const unsigned short* bXl = sb[buf][1];
    const unsigned short* bWh = sb[buf][2];
    const unsigned short* bWl = sb[buf][3];

    // B fragments (32x16 K x N): lane = N, elems = contiguous K half
    v16bf bh[2], bl[2];
#pragma unroll
    for (int ni = 0; ni < 2; ++ni) {
      const int n = wc * 32 + ni * 16 + lm;
      const int o = n * LSTR + lh * 16;
      bh[ni] = ld_frag2(&bWh[o], &bWh[o + 8]);
      bl[ni] = ld_frag2(&bWl[o], &bWl[o + 8]);
    }

    // A fragments (16x32 M x K) + 3-product split-bf16 WMMA
#pragma unroll
    for (int mi = 0; mi < 4; ++mi) {
      const int m = wr * 64 + mi * 16 + lm;
      const int o = m * LSTR + lh * 16;
      v16bf ah = ld_frag2(&bXh[o], &bXh[o + 8]);
      v16bf al = ld_frag2(&bXl[o], &bXl[o + 8]);
#pragma unroll
      for (int ni = 0; ni < 2; ++ni) {
        acc[mi][ni] = __builtin_amdgcn_wmma_f32_16x16x32_bf16(
            false, ah, false, bh[ni], (short)0, acc[mi][ni], false, false);
        acc[mi][ni] = __builtin_amdgcn_wmma_f32_16x16x32_bf16(
            false, ah, false, bl[ni], (short)0, acc[mi][ni], false, false);
        acc[mi][ni] = __builtin_amdgcn_wmma_f32_16x16x32_bf16(
            false, al, false, bh[ni], (short)0, acc[mi][ni], false, false);
      }
    }

    // ---- drop next tile into the alternate buffer ----
    if (more) {
      const int nb = buf ^ 1;
      *(v8us*)&sb[nb][0][so]     = st[0];
      *(v8us*)&sb[nb][0][so + 8] = st[1];
      *(v8us*)&sb[nb][1][so]     = st[2];
      *(v8us*)&sb[nb][1][so + 8] = st[3];
      *(v8us*)&sb[nb][2][so]     = st[4];
      *(v8us*)&sb[nb][2][so + 8] = st[5];
      *(v8us*)&sb[nb][3][so]     = st[6];
      *(v8us*)&sb[nb][3][so + 8] = st[7];
    }
    __syncthreads();
    buf ^= 1;
  }

  // epilogue: bias + store (C/D: lane = N, VGPR r -> M = 8*lh + r)
#pragma unroll
  for (int ni = 0; ni < 2; ++ni) {
    const int col = bn * 128 + wc * 32 + ni * 16 + lm;
    const float bvv = Bv[col];
#pragma unroll
    for (int mi = 0; mi < 4; ++mi) {
      const int row0 = bm * 128 + wr * 64 + mi * 16 + lh * 8;
#pragma unroll
      for (int r = 0; r < 8; ++r) {
        O[(size_t)(row0 + r) * OUT_F + col] = acc[mi][ni][r] + bvv;
      }
    }
  }
}

// ---------------- Fallback: fused split+GEMM (if workspace too small) ------
__global__ __launch_bounds__(256)
void linear_wmma_fused(const float* __restrict__ X,
                       const float* __restrict__ W,
                       const float* __restrict__ Bv,
                       float* __restrict__ O) {
  __shared__ unsigned short sXhi[128 * LSTR];
  __shared__ unsigned short sXlo[128 * LSTR];
  __shared__ unsigned short sWhi[128 * LSTR];
  __shared__ unsigned short sWlo[128 * LSTR];

  const int bn = blockIdx.x;
  const int bm = blockIdx.y;
  const int t    = threadIdx.x;
  const int lane = t & 31;
  const int wave = t >> 5;
  const int wr = wave & 1;
  const int wc = wave >> 1;
  const int lm = lane & 15;
  const int lh = lane >> 4;

  const int srow = t >> 1;
  const int scol = (t & 1) * 16;
  const float* xg = X + (size_t)(bm * 128 + srow) * IN_F + scol;
  const float* wg = W + (size_t)(bn * 128 + srow) * IN_F + scol;

  v8f acc[4][2];
#pragma unroll
  for (int mi = 0; mi < 4; ++mi)
#pragma unroll
    for (int ni = 0; ni < 2; ++ni)
#pragma unroll
      for (int r = 0; r < 8; ++r) acc[mi][ni][r] = 0.0f;

  for (int k0 = 0; k0 < IN_F; k0 += TILE_K) {
    const float4* x4 = (const float4*)(xg + k0);
    const float4* w4 = (const float4*)(wg + k0);
#pragma unroll
    for (int c = 0; c < 4; ++c) {
      float4 xv = x4[c];
      float4 wv = w4[c];
      float xf[4] = {xv.x, xv.y, xv.z, xv.w};
      float wf[4] = {wv.x, wv.y, wv.z, wv.w};
      v4us xh, xl, wh, wl;
#pragma unroll
      for (int j = 0; j < 4; ++j) {
        unsigned short h = f2bf(xf[j]);
        xh[j] = h;
        xl[j] = f2bf(xf[j] - bf2f(h));
        unsigned short g = f2bf(wf[j]);
        wh[j] = g;
        wl[j] = f2bf(wf[j] - bf2f(g));
      }
      const int o = srow * LSTR + scol + c * 4;
      *(v4us*)&sXhi[o] = xh;
      *(v4us*)&sXlo[o] = xl;
      *(v4us*)&sWhi[o] = wh;
      *(v4us*)&sWlo[o] = wl;
    }
    __syncthreads();

    v16bf bh[2], bl[2];
#pragma unroll
    for (int ni = 0; ni < 2; ++ni) {
      const int n = wc * 32 + ni * 16 + lm;
      const int o = n * LSTR + lh * 16;
      bh[ni] = ld_frag2(&sWhi[o], &sWhi[o + 8]);
      bl[ni] = ld_frag2(&sWlo[o], &sWlo[o + 8]);
    }
#pragma unroll
    for (int mi = 0; mi < 4; ++mi) {
      const int m = wr * 64 + mi * 16 + lm;
      const int o = m * LSTR + lh * 16;
      v16bf ah = ld_frag2(&sXhi[o], &sXhi[o + 8]);
      v16bf al = ld_frag2(&sXlo[o], &sXlo[o + 8]);
#pragma unroll
      for (int ni = 0; ni < 2; ++ni) {
        acc[mi][ni] = __builtin_amdgcn_wmma_f32_16x16x32_bf16(
            false, ah, false, bh[ni], (short)0, acc[mi][ni], false, false);
        acc[mi][ni] = __builtin_amdgcn_wmma_f32_16x16x32_bf16(
            false, ah, false, bl[ni], (short)0, acc[mi][ni], false, false);
        acc[mi][ni] = __builtin_amdgcn_wmma_f32_16x16x32_bf16(
            false, al, false, bh[ni], (short)0, acc[mi][ni], false, false);
      }
    }
    __syncthreads();
  }

#pragma unroll
  for (int ni = 0; ni < 2; ++ni) {
    const int col = bn * 128 + wc * 32 + ni * 16 + lm;
    const float bvv = Bv[col];
#pragma unroll
    for (int mi = 0; mi < 4; ++mi) {
      const int row0 = bm * 128 + wr * 64 + mi * 16 + lh * 8;
#pragma unroll
      for (int r = 0; r < 8; ++r) {
        O[(size_t)(row0 + r) * OUT_F + col] = acc[mi][ni][r] + bvv;
      }
    }
  }
}

extern "C" void kernel_launch(void* const* d_in, const int* in_sizes, int n_in,
                              void* d_out, int out_size, void* d_ws, size_t ws_size,
                              hipStream_t stream) {
  const float* x = (const float*)d_in[0];      // [4096, 2048]
  const float* w = (const float*)d_in[1];      // [2048, 2048]
  const float* b = (const float*)d_in[2];      // [2048]
  float* out = (float*)d_out;                  // [4096, 2048]

  const size_t nX = (size_t)BATCHN * IN_F;     // 8M elems
  const size_t nW = (size_t)OUT_F * IN_F;      // 4M elems
  const size_t need = (nX + nW) * 2 * sizeof(unsigned short) * 2 / 2; // hi+lo for X and W
  dim3 grid(OUT_F / 128, BATCHN / 128);        // 16 x 32
  dim3 block(256);

  if (ws_size >= (nX + nW) * 4) {              // 48 MB split workspace
    unsigned short* xhi = (unsigned short*)d_ws;
    unsigned short* xlo = xhi + nX;
    unsigned short* whi = xlo + nX;
    unsigned short* wlo = whi + nW;
    split_pass<<<(int)(nX / 4 / 256), 256, 0, stream>>>(x, xhi, xlo, (int)(nX / 4));
    split_pass<<<(int)(nW / 4 / 256), 256, 0, stream>>>(w, whi, wlo, (int)(nW / 4));
    gemm_split<<<grid, block, 0, stream>>>(xhi, xlo, whi, wlo, b, out);
  } else {
    linear_wmma_fused<<<grid, block, 0, stream>>>(x, w, b, out);
  }
  (void)need;
}
